// FinalGraphTransformerModule_26482768347808
// MI455X (gfx1250) — matrix operand order repaired and checked by reading, not compiled, not run on hardware
//
#include <hip/hip_runtime.h>
#include <hip/hip_bf16.h>

typedef __attribute__((ext_vector_type(16))) _Float16 v16h;
typedef __attribute__((ext_vector_type(8)))  _Float16 v8h;
typedef __attribute__((ext_vector_type(8)))  float    v8f;

#define DEV static __device__ __forceinline__

static constexpr int   NN = 50000;
static constexpr int   EE = 600000;
static constexpr int   DD = 128;
static constexpr int   HH = 4;
static constexpr int   GG = 64;

DEV float wave_sum(float v) {
#pragma unroll
  for (int m = 16; m >= 1; m >>= 1) v += __shfl_xor(v, m, 32);
  return v;
}

DEV unsigned f2ord(float f) {
  unsigned u = __float_as_uint(f);
  return (u & 0x80000000u) ? ~u : (u | 0x80000000u);
}
DEV float ord2f(unsigned u) {
  return (u & 0x80000000u) ? __uint_as_float(u ^ 0x80000000u) : __uint_as_float(~u);
}
DEV float sigm(float x) { return 1.0f / (1.0f + expf(-x)); }

// ---------------- weight transpose + cast: Wt[c][r] = (f16)W[r][c] ----------
__global__ void k_transpose_cast(const float* __restrict__ W, _Float16* __restrict__ Wt,
                                 int rows, int cols) {
  int i = blockIdx.x * blockDim.x + threadIdx.x;
  if (i >= rows * cols) return;
  int r = i / cols, c = i - r * cols;
  Wt[(size_t)c * rows + r] = (_Float16)W[i];
}

// ---------------- layernorm over rows of 128, f16 output --------------------
__global__ __launch_bounds__(256) void k_layernorm_cast(
    const float* __restrict__ X, const float* __restrict__ w, const float* __restrict__ b,
    _Float16* __restrict__ Y, int nrows) {
  int wid  = (blockIdx.x * blockDim.x + threadIdx.x) >> 5;
  int lane = threadIdx.x & 31;
  if (wid >= nrows) return;
  const float4 vv = *(const float4*)(X + (size_t)wid * DD + lane * 4);
  float v[4] = {vv.x, vv.y, vv.z, vv.w};
  float mean = wave_sum(v[0] + v[1] + v[2] + v[3]) * (1.0f / 128.0f);
  float d[4], sq = 0.f;
#pragma unroll
  for (int i = 0; i < 4; ++i) { d[i] = v[i] - mean; sq += d[i] * d[i]; }
  float inv = rsqrtf(wave_sum(sq) * (1.0f / 128.0f) + 1e-5f);
  _Float16* y = Y + (size_t)wid * DD + lane * 4;
#pragma unroll
  for (int i = 0; i < 4; ++i)
    y[i] = (_Float16)(d[i] * inv * w[lane * 4 + i] + b[lane * 4 + i]);
}

// ---------------- WMMA GEMM: C[M x Nc] = A_f16[M x K] * Bt_f16[Nc x K] ------
// wave per 16-row tile; A frags contiguous (k = g*16 + s, as for B),
// B frags are contiguous (weights pre-transposed).
template <int K, bool HOUT>
__global__ __launch_bounds__(256) void k_wmma_gemm(
    const _Float16* __restrict__ A, const _Float16* __restrict__ Bt,
    void* __restrict__ Cv, int M, int Ncols) {
  constexpr int KT = K / 32;
  int wid  = (blockIdx.x * blockDim.x + threadIdx.x) >> 5;
  int lane = threadIdx.x & 31;
  if (wid >= (M >> 4)) return;                       // wave-uniform: EXEC stays full
  int m = lane & 15, g = lane >> 4;

  const _Float16* arow = A + (size_t)(wid * 16 + m) * K + g * 16;
  v16h a[KT];
#pragma unroll
  for (int kk = 0; kk < KT; ++kk) {
    v8h lo = *(const v8h*)(arow + kk * 32);          // K = g*16 .. g*16+7
    v8h hi = *(const v8h*)(arow + kk * 32 + 8);      // K = g*16+8 .. g*16+15
    v16h t;
#pragma unroll
    for (int i = 0; i < 8; ++i) { t[i] = lo[i]; t[8 + i] = hi[i]; }
    a[kk] = t;
  }

  int nColTiles = Ncols >> 4;
  for (int ct = 0; ct < nColTiles; ++ct) {
    const _Float16* bcol = Bt + (size_t)(ct * 16 + m) * K + g * 16;
    if (ct + 1 < nColTiles) __builtin_prefetch(bcol + 16 * K, 0, 1);
    v8f acc = {};
#pragma unroll
    for (int kk = 0; kk < KT; ++kk) {
      v16h bf = *(const v16h*)(bcol + kk * 32);      // K = g*16 .. g*16+15
      acc = __builtin_amdgcn_wmma_f32_16x16x32_f16(false, a[kk], false, bf,
                                                   (short)0, acc, false, false);
    }
    size_t rbase = (size_t)(wid * 16 + g * 8) * Ncols + ct * 16 + m;
    if (HOUT) {
      _Float16* C = (_Float16*)Cv;
#pragma unroll
      for (int r = 0; r < 8; ++r) C[rbase + (size_t)r * Ncols] = (_Float16)acc[r];
    } else {
      float* C = (float*)Cv;
#pragma unroll
      for (int r = 0; r < 8; ++r) C[rbase + (size_t)r * Ncols] = acc[r];
    }
  }
}

// ---------------- edge attention: wave per (edge, head) ---------------------
__global__ __launch_bounds__(256) void k_edge_attn(
    const float* __restrict__ Q, const float* __restrict__ Kf, const float* __restrict__ V,
    const _Float16* __restrict__ Pe, const int* __restrict__ src, const int* __restrict__ dst,
    float* __restrict__ wV, float* __restrict__ z) {
  int wid  = (blockIdx.x * blockDim.x + threadIdx.x) >> 5;
  int lane = threadIdx.x & 31;
  int e = wid >> 2, h = wid & 3;
  if (e >= EE) return;
  int s = src[e], d = dst[e];
  int c = h * 32 + lane;
  float q  = Q [(size_t)d * DD + c];
  float k  = Kf[(size_t)s * DD + c];
  float pe = (float)Pe[(size_t)e * DD + c];
  float sc = k * q * 0.17677669529663687f;           // 1/sqrt(32)
  sc = fminf(5.0f, fmaxf(-5.0f, sc)) * pe;
  float tot = fminf(5.0f, fmaxf(-5.0f, wave_sum(sc)));
  float sexp = expf(tot);
  float v = V[(size_t)s * DD + c];
  atomicAdd(&wV[(size_t)d * DD + c], v * sexp);
  if (lane == 0) atomicAdd(&z[(size_t)d * HH + h], sexp);
}

// ---------------- elementwise helpers ---------------------------------------
__global__ void k_zero_f32(float* p, long long n) {
  long long i = (long long)blockIdx.x * blockDim.x + threadIdx.x;
  if (i < n) p[i] = 0.0f;
}
__global__ void k_attn_norm_cast(const float* __restrict__ wV, const float* __restrict__ z,
                                 _Float16* __restrict__ out, long long n) {
  long long i = (long long)blockIdx.x * blockDim.x + threadIdx.x;
  if (i >= n) return;
  int node = (int)(i >> 7), c = (int)(i & 127);
  out[i] = (_Float16)(wV[i] / (z[node * HH + (c >> 5)] + 1e-6f));
}
__global__ void k_bias_residual(const float* __restrict__ Cin, const float* __restrict__ bo,
                                const float* __restrict__ resid, float* __restrict__ x,
                                long long n) {
  long long i = (long long)blockIdx.x * blockDim.x + threadIdx.x;
  if (i < n) x[i] = resid[i] + Cin[i] + bo[i & 127];
}
__global__ void k_silu_cast(const float* __restrict__ t1, _Float16* __restrict__ out, long long n) {
  long long i = (long long)blockIdx.x * blockDim.x + threadIdx.x;
  if (i >= n) return;
  float v = t1[i];
  out[i] = (_Float16)(v * sigm(v));
}
__global__ void k_add(const float* __restrict__ a, const float* __restrict__ b,
                      float* __restrict__ o, long long n) {
  long long i = (long long)blockIdx.x * blockDim.x + threadIdx.x;
  if (i < n) o[i] = a[i] + b[i];
}

// ---------------- Set2Set ----------------------------------------------------
__global__ void k_s2s_init(float* qstar, float* hh, float* cc) {
  int i = blockIdx.x * blockDim.x + threadIdx.x;
  if (i < GG * 2 * DD) qstar[i] = 0.0f;
  if (i < GG * DD) { hh[i] = 0.0f; cc[i] = 0.0f; }
}
__global__ void k_iter_init(unsigned* m, float* denom, float* ro) {
  int i = blockIdx.x * blockDim.x + threadIdx.x;
  if (i < GG) { m[i] = 0u; denom[i] = 0.0f; }
  if (i < GG * DD) ro[i] = 0.0f;
}
__global__ void k_lstm_gates(const float* __restrict__ qstar, const float* __restrict__ hh,
                             const float* __restrict__ Wih, const float* __restrict__ Whh,
                             const float* __restrict__ bih, const float* __restrict__ bhh,
                             float* __restrict__ gates) {
  int i = blockIdx.x * blockDim.x + threadIdx.x;
  if (i >= GG * 4 * DD) return;
  int g = i >> 9, j = i & 511;
  float acc = bih[j] + bhh[j];
  const float* wr = Wih + (size_t)j * (2 * DD);
  const float* qs = qstar + g * (2 * DD);
  for (int t = 0; t < 2 * DD; ++t) acc += qs[t] * wr[t];
  const float* wr2 = Whh + (size_t)j * DD;
  const float* hr  = hh + g * DD;
  for (int t = 0; t < DD; ++t) acc += hr[t] * wr2[t];
  gates[i] = acc;
}
__global__ void k_lstm_update(const float* __restrict__ gates, float* __restrict__ hh,
                              float* __restrict__ cc) {
  int i = blockIdx.x * blockDim.x + threadIdx.x;
  if (i >= GG * DD) return;
  int g = i >> 7, d = i & 127;
  const float* gr = gates + g * 512;
  float ig = gr[d], fg = gr[128 + d], gg = gr[256 + d], og = gr[384 + d];
  float c = sigm(fg) * cc[i] + sigm(ig) * tanhf(gg);
  cc[i] = c;
  hh[i] = sigm(og) * tanhf(c);
}
__global__ __launch_bounds__(256) void k_ener(
    const float* __restrict__ nf, const float* __restrict__ hh, const int* __restrict__ gid,
    float* __restrict__ ener, unsigned* __restrict__ m) {
  int wid = (blockIdx.x * blockDim.x + threadIdx.x) >> 5;
  int lane = threadIdx.x & 31;
  if (wid >= NN) return;
  int g = gid[wid];
  float4 a = *(const float4*)(nf + (size_t)wid * DD + lane * 4);
  float4 q = *(const float4*)(hh + (size_t)g * DD + lane * 4);
  float s = wave_sum(a.x * q.x + a.y * q.y + a.z * q.z + a.w * q.w);
  if (lane == 0) { ener[wid] = s; atomicMax(&m[g], f2ord(s)); }
}
__global__ void k_exp_denom(const float* __restrict__ ener, const unsigned* __restrict__ m,
                            const int* __restrict__ gid, float* __restrict__ ex,
                            float* __restrict__ denom) {
  int i = blockIdx.x * blockDim.x + threadIdx.x;
  if (i >= NN) return;
  int g = gid[i];
  float e = expf(ener[i] - ord2f(m[g]));
  ex[i] = e;
  atomicAdd(&denom[g], e);
}
__global__ void k_readout(const float* __restrict__ nf, const float* __restrict__ ex,
                          const float* __restrict__ denom, const int* __restrict__ gid,
                          float* __restrict__ ro, long long n) {
  long long i = (long long)blockIdx.x * blockDim.x + threadIdx.x;
  if (i >= n) return;
  int node = (int)(i >> 7), d = (int)(i & 127);
  int g = gid[node];
  atomicAdd(&ro[g * DD + d], (ex[node] / denom[g]) * nf[i]);
}
__global__ void k_qstar(const float* __restrict__ hh, const float* __restrict__ ro,
                        float* __restrict__ qstar) {
  int i = blockIdx.x * blockDim.x + threadIdx.x;
  if (i >= GG * DD) return;
  int g = i >> 7, d = i & 127;
  qstar[g * 2 * DD + d] = hh[i];
  qstar[g * 2 * DD + DD + d] = ro[i];
}

// ---------------- host orchestration ----------------------------------------
extern "C" void kernel_launch(void* const* d_in, const int* in_sizes, int n_in,
                              void* d_out, int out_size, void* d_ws, size_t ws_size,
                              hipStream_t stream) {
  (void)in_sizes; (void)n_in; (void)out_size; (void)ws_size;
  const float* node = (const float*)d_in[0];
  const float* edge = (const float*)d_in[1];
  const float* Wq = (const float*)d_in[2];
  const float* Wk = (const float*)d_in[3];
  const float* Wv = (const float*)d_in[4];
  const float* We = (const float*)d_in[5];
  const float* Wo = (const float*)d_in[6];
  const float* bo = (const float*)d_in[7];
  const float* W1 = (const float*)d_in[8];
  const float* W2 = (const float*)d_in[9];
  const float* ln1nw = (const float*)d_in[10];
  const float* ln1nb = (const float*)d_in[11];
  const float* ln1ew = (const float*)d_in[12];
  const float* ln1eb = (const float*)d_in[13];
  const float* ln2w = (const float*)d_in[14];
  const float* ln2b = (const float*)d_in[15];
  const float* Wih = (const float*)d_in[16];
  const float* Whh = (const float*)d_in[17];
  const float* bih = (const float*)d_in[18];
  const float* bhh = (const float*)d_in[19];
  const int* src = (const int*)d_in[20];
  const int* dst = (const int*)d_in[21];
  const int* gid = (const int*)d_in[22];

  char* ws = (char*)d_ws;
  size_t off = 0;
  auto alloc = [&](size_t bytes) -> char* {
    char* p = ws + off;
    off = (off + bytes + 255) & ~(size_t)255;
    return p;
  };
  _Float16* h16   = (_Float16*)alloc((size_t)NN * DD * 2);
  _Float16* e16   = (_Float16*)alloc((size_t)EE * DD * 2);
  _Float16* pe16  = (_Float16*)alloc((size_t)EE * DD * 2);
  _Float16* wqT = (_Float16*)alloc(DD * DD * 2);
  _Float16* wkT = (_Float16*)alloc(DD * DD * 2);
  _Float16* wvT = (_Float16*)alloc(DD * DD * 2);
  _Float16* weT = (_Float16*)alloc(DD * DD * 2);
  _Float16* woT = (_Float16*)alloc(DD * DD * 2);
  _Float16* w1T = (_Float16*)alloc(DD * 2 * DD * 2);
  _Float16* w2T = (_Float16*)alloc(2 * DD * DD * 2);
  float* Qb  = (float*)alloc((size_t)NN * DD * 4);
  float* Kb  = (float*)alloc((size_t)NN * DD * 4);
  float* Vb  = (float*)alloc((size_t)NN * DD * 4);
  float* wVb = (float*)alloc((size_t)NN * DD * 4);
  float* zb  = (float*)alloc((size_t)NN * HH * 4);
  _Float16* attn16 = (_Float16*)alloc((size_t)NN * DD * 2);
  float* aoC = (float*)alloc((size_t)NN * DD * 4);
  float* xb  = (float*)alloc((size_t)NN * DD * 4);
  _Float16* y16 = (_Float16*)alloc((size_t)NN * DD * 2);
  float* t1  = (float*)alloc((size_t)NN * 2 * DD * 4);
  _Float16* t1h = (_Float16*)alloc((size_t)NN * 2 * DD * 2);
  float* t2  = (float*)alloc((size_t)NN * DD * 4);
  float* nf  = (float*)alloc((size_t)NN * DD * 4);
  float* ener = (float*)alloc((size_t)NN * 4);
  float* exb  = (float*)alloc((size_t)NN * 4);
  float* hhb  = (float*)alloc(GG * DD * 4);
  float* ccb  = (float*)alloc(GG * DD * 4);
  float* gatesb = (float*)alloc(GG * 4 * DD * 4);
  unsigned* mb = (unsigned*)alloc(GG * 4);
  float* denb  = (float*)alloc(GG * 4);
  float* rob   = (float*)alloc(GG * DD * 4);
  float* qstar = (float*)d_out;                        // [G, 2D] output

  const int TB = 256;
  auto ceil_div = [](long long a, long long b) { return (int)((a + b - 1) / b); };

  // weights -> f16 transposed
  k_transpose_cast<<<ceil_div(DD * DD, TB), TB, 0, stream>>>(Wq, wqT, DD, DD);
  k_transpose_cast<<<ceil_div(DD * DD, TB), TB, 0, stream>>>(Wk, wkT, DD, DD);
  k_transpose_cast<<<ceil_div(DD * DD, TB), TB, 0, stream>>>(Wv, wvT, DD, DD);
  k_transpose_cast<<<ceil_div(DD * DD, TB), TB, 0, stream>>>(We, weT, DD, DD);
  k_transpose_cast<<<ceil_div(DD * DD, TB), TB, 0, stream>>>(Wo, woT, DD, DD);
  k_transpose_cast<<<ceil_div(DD * 2 * DD, TB), TB, 0, stream>>>(W1, w1T, DD, 2 * DD);
  k_transpose_cast<<<ceil_div(2 * DD * DD, TB), TB, 0, stream>>>(W2, w2T, 2 * DD, DD);

  // pre-norms
  k_layernorm_cast<<<ceil_div((long long)NN * 32, TB), TB, 0, stream>>>(node, ln1nw, ln1nb, h16, NN);
  k_layernorm_cast<<<ceil_div((long long)EE * 32, TB), TB, 0, stream>>>(edge, ln1ew, ln1eb, e16, EE);

  // projections (WMMA)
  int gN = ceil_div(NN / 16, 8), gE = ceil_div(EE / 16, 8);
  k_wmma_gemm<128, false><<<gN, TB, 0, stream>>>(h16, wqT, (void*)Qb, NN, DD);
  k_wmma_gemm<128, false><<<gN, TB, 0, stream>>>(h16, wkT, (void*)Kb, NN, DD);
  k_wmma_gemm<128, false><<<gN, TB, 0, stream>>>(h16, wvT, (void*)Vb, NN, DD);
  k_wmma_gemm<128, true ><<<gE, TB, 0, stream>>>(e16, weT, (void*)pe16, EE, DD);

  // edge attention + scatter aggregation
  k_zero_f32<<<ceil_div((long long)NN * DD, TB), TB, 0, stream>>>(wVb, (long long)NN * DD);
  k_zero_f32<<<ceil_div((long long)NN * HH, TB), TB, 0, stream>>>(zb, (long long)NN * HH);
  k_edge_attn<<<ceil_div((long long)EE * HH * 32, TB), TB, 0, stream>>>(Qb, Kb, Vb, pe16, src, dst, wVb, zb);

  // attn normalize -> f16, output proj, bias + residual
  k_attn_norm_cast<<<ceil_div((long long)NN * DD, TB), TB, 0, stream>>>(wVb, zb, attn16, (long long)NN * DD);
  k_wmma_gemm<128, false><<<gN, TB, 0, stream>>>(attn16, woT, (void*)aoC, NN, DD);
  k_bias_residual<<<ceil_div((long long)NN * DD, TB), TB, 0, stream>>>(aoC, bo, node, xb, (long long)NN * DD);

  // FFN
  k_layernorm_cast<<<ceil_div((long long)NN * 32, TB), TB, 0, stream>>>(xb, ln2w, ln2b, y16, NN);
  k_wmma_gemm<128, false><<<gN, TB, 0, stream>>>(y16, w1T, (void*)t1, NN, 2 * DD);
  k_silu_cast<<<ceil_div((long long)NN * 2 * DD, TB), TB, 0, stream>>>(t1, t1h, (long long)NN * 2 * DD);
  k_wmma_gemm<256, false><<<gN, TB, 0, stream>>>(t1h, w2T, (void*)t2, NN, DD);
  k_add<<<ceil_div((long long)NN * DD, TB), TB, 0, stream>>>(xb, t2, nf, (long long)NN * DD);

  // Set2Set readout (3 iterations)
  k_s2s_init<<<ceil_div(GG * 2 * DD, TB), TB, 0, stream>>>(qstar, hhb, ccb);
  for (int it = 0; it < 3; ++it) {
    k_lstm_gates<<<ceil_div(GG * 4 * DD, TB), TB, 0, stream>>>(qstar, hhb, Wih, Whh, bih, bhh, gatesb);
    k_lstm_update<<<ceil_div(GG * DD, TB), TB, 0, stream>>>(gatesb, hhb, ccb);
    k_iter_init<<<ceil_div(GG * DD, TB), TB, 0, stream>>>(mb, denb, rob);
    k_ener<<<ceil_div((long long)NN * 32, TB), TB, 0, stream>>>(nf, hhb, gid, ener, mb);
    k_exp_denom<<<ceil_div(NN, TB), TB, 0, stream>>>(ener, mb, gid, exb, denb);
    k_readout<<<ceil_div((long long)NN * DD, TB), TB, 0, stream>>>(nf, exb, denb, gid, rob, (long long)NN * DD);
    k_qstar<<<ceil_div(GG * DD, TB), TB, 0, stream>>>(hhb, rob, qstar);
  }
}
